// TabAMLModel_30039001268723
// MI455X (gfx1250) — matrix operand order repaired and checked by reading, not compiled, not run on hardware
//
#include <hip/hip_runtime.h>
#include <hip/hip_bf16.h>
#include <math.h>

// ---------------- model constants ----------------
#define BB      16384
#define NCAT    16
#define DD      64
#define HH      4
#define DH      16
#define SHAREDN 10
#define INDIV   54
#define CONTN   16
#define LL      2
#define FLATN   (NCAT*DD + CONTN)      // 1040
#define N1      (4*FLATN)              // 4160
#define N2      (2*FLATN)              // 2080
#define EPSLN   1e-5f

#define MB      2048                   // MLP row batch
#define NBATCH  (BB/MB)                // 8
#define KC1     33                     // ceil(1040/32)
#define NT1     (N1/16)                // 260
#define KC2     (N1/32)                // 130
#define NT2     (N2/16)                // 130

typedef __attribute__((ext_vector_type(16))) _Float16 v16h;
typedef __attribute__((ext_vector_type(8)))  float    v8f;

__device__ __forceinline__ float gelu_exact(float x) {
    return 0.5f * x * (1.0f + erff(x * 0.70710678118654752440f));
}

#define WMMA_F16(A, Bf, Cacc) \
    __builtin_amdgcn_wmma_f32_16x16x32_f16(false, (A), false, (Bf), (short)0, (Cacc), false, false)

// Build 16-bit A fragment (16 rows x 32 K) for V_WMMA_F32_16X16X32_F16 from f32 rows.
// Lane l holds row m = l&15; khalf = l>>4. Element e maps to
// K = khalf*8 + e            (e < 8)
// K = 16 + khalf*8 + (e-8)   (e >= 8)
__device__ __forceinline__ v16h load_a_frag(const float* __restrict__ A,
                                            int lda, int k0, int K, int lane) {
    int m  = lane & 15;
    int kh = lane >> 4;
    const float* rp = A + (size_t)m * lda;
    v16h a;
#pragma unroll
    for (int e = 0; e < 8; ++e) {
        int k1 = k0 + kh * 8 + e;
        int k2 = k0 + 16 + kh * 8 + e;
        a[e]     = (_Float16)((k1 < K) ? rp[k1] : 0.0f);
        a[8 + e] = (_Float16)((k2 < K) ? rp[k2] : 0.0f);
    }
    return a;
}

// Accumulate a 16x16 tile: A[16xK] (f32, row stride lda) x packed W frags (LDS path).
__device__ __forceinline__ v8f wmma_tile(const float* __restrict__ A, int lda,
                                         int K, int KC,
                                         const v16h* __restrict__ wfrags,
                                         int lane, v8f acc) {
    for (int kc = 0; kc < KC; ++kc) {
        v16h af = load_a_frag(A, lda, kc * 32, K, lane);
        v16h bf = wfrags[kc * 32 + lane];
        acc = WMMA_F16(af, bf, acc);
    }
    return acc;
}

// ---------------- kernel 1: embedding gather ----------------
__global__ void build_tokens_kernel(const int* __restrict__ x_cat,
                                    const float* __restrict__ shared_emb,
                                    const float* __restrict__ emb_tables,
                                    float* __restrict__ tokens) {
    int idx = blockIdx.x * blockDim.x + threadIdx.x;   // over B*NCAT*D
    int d = idx & 63;
    int c = (idx >> 6) & 15;
    int b = idx >> 10;
    float v;
    if (d < SHAREDN) {
        v = shared_emb[d];
    } else {
        int cat = x_cat[b * NCAT + c];
        v = emb_tables[((size_t)c * 1000 + cat) * INDIV + (d - SHAREDN)];
    }
    tokens[idx] = v;
}

// ---------------- kernel 2: micro encoder (t=2), one wave per sample ----------------
__global__ __launch_bounds__(256)
void micro_encoder_kernel(float* __restrict__ tokens,
                          const float* __restrict__ qkvw,
                          const float* __restrict__ outw,
                          const float* __restrict__ ln1w, const float* __restrict__ ln1b,
                          const float* __restrict__ ln2w, const float* __restrict__ ln2b,
                          const float* __restrict__ w1,  const float* __restrict__ b1,
                          const float* __restrict__ w2,  const float* __restrict__ b2) {
    __shared__ float sm[8][1184];
    int wave = threadIdx.x >> 5;
    int lane = threadIdx.x & 31;
    int s = blockIdx.x * 8 + wave;

    float* xs = sm[wave];          // [2][64]
    float* xn = xs + 128;          // [2][64]
    float* qk = xs + 256;          // [2][192]
    float* hb = xs + 640;          // [2][256]  (also o-buffer [2][64])
    float* ps = xs + 1152;         // [4][2][2] prev scores

#pragma unroll
    for (int i = 0; i < 4; ++i) xs[lane + 32 * i] = tokens[(size_t)s * 1024 + lane + 32 * i];
    if (lane < 16) ps[lane] = 0.0f;
    __syncthreads();

    for (int l = 0; l < LL; ++l) {
        // --- LN1 ---
        if (lane < 2) {
            int t = lane;
            float m = 0.f;
            for (int d = 0; d < 64; ++d) m += xs[t * 64 + d];
            m *= (1.0f / 64.0f);
            float var = 0.f;
            for (int d = 0; d < 64; ++d) { float df = xs[t * 64 + d] - m; var += df * df; }
            var *= (1.0f / 64.0f);
            float inv = 1.0f / sqrtf(var + EPSLN);
            for (int d = 0; d < 64; ++d)
                xn[t * 64 + d] = (xs[t * 64 + d] - m) * inv * ln1w[l * 64 + d] + ln1b[l * 64 + d];
        }
        __syncthreads();
        // --- qkv = xn @ qw  (2x192) ---
        for (int i = 0; i < 12; ++i) {
            int idx = lane + 32 * i;      // 0..383
            int t = idx / 192, o = idx % 192;
            float acc = 0.f;
            for (int k = 0; k < 64; ++k)
                acc += xn[t * 64 + k] * qkvw[(size_t)l * 12288 + k * 192 + o];
            qk[t * 192 + o] = acc;
        }
        __syncthreads();
        // --- attention (t=2) ---
        if (lane < 8) {
            int h = lane >> 1, t = lane & 1;
            float sc[2];
            for (int j = 0; j < 2; ++j) {
                float d = 0.f;
                for (int dd = 0; dd < 16; ++dd)
                    d += qk[t * 192 + h * 16 + dd] * qk[j * 192 + 64 + h * 16 + dd];
                sc[j] = d * 0.25f + ps[(h * 2 + t) * 2 + j];
                ps[(h * 2 + t) * 2 + j] = sc[j];   // carry raw scores to next layer
            }
            float mx = fmaxf(sc[0], sc[1]);
            float e0 = expf(sc[0] - mx), e1 = expf(sc[1] - mx);
            float inv = 1.0f / (e0 + e1);
            e0 *= inv; e1 *= inv;
            for (int dd = 0; dd < 16; ++dd)
                hb[t * 64 + h * 16 + dd] = e0 * qk[0 * 192 + 128 + h * 16 + dd]
                                         + e1 * qk[1 * 192 + 128 + h * 16 + dd];
        }
        __syncthreads();
        // --- out proj + residual ---
        for (int i = 0; i < 4; ++i) {
            int idx = lane + 32 * i;     // 0..127
            int t = idx >> 6, d = idx & 63;
            float acc = 0.f;
            for (int k = 0; k < 64; ++k)
                acc += hb[t * 64 + k] * outw[(size_t)l * 4096 + k * 64 + d];
            xs[idx] += acc;
        }
        __syncthreads();
        // --- LN2 ---
        if (lane < 2) {
            int t = lane;
            float m = 0.f;
            for (int d = 0; d < 64; ++d) m += xs[t * 64 + d];
            m *= (1.0f / 64.0f);
            float var = 0.f;
            for (int d = 0; d < 64; ++d) { float df = xs[t * 64 + d] - m; var += df * df; }
            var *= (1.0f / 64.0f);
            float inv = 1.0f / sqrtf(var + EPSLN);
            for (int d = 0; d < 64; ++d)
                xn[t * 64 + d] = (xs[t * 64 + d] - m) * inv * ln2w[l * 64 + d] + ln2b[l * 64 + d];
        }
        __syncthreads();
        // --- ff1 + gelu ---
        for (int i = 0; i < 16; ++i) {
            int idx = lane + 32 * i;     // 0..511
            int t = idx >> 8, o = idx & 255;
            float acc = b1[l * 256 + o];
            for (int k = 0; k < 64; ++k)
                acc += xn[t * 64 + k] * w1[(size_t)l * 16384 + k * 256 + o];
            hb[t * 256 + o] = gelu_exact(acc);
        }
        __syncthreads();
        // --- ff2 + residual ---
        for (int i = 0; i < 4; ++i) {
            int idx = lane + 32 * i;
            int t = idx >> 6, d = idx & 63;
            float acc = b2[l * 64 + d];
            for (int k = 0; k < 256; ++k)
                acc += hb[t * 256 + k] * w2[(size_t)l * 16384 + k * 64 + d];
            xs[idx] += acc;
        }
        __syncthreads();
    }
#pragma unroll
    for (int i = 0; i < 4; ++i) tokens[(size_t)s * 1024 + lane + 32 * i] = xs[lane + 32 * i];
}

// ---------------- kernel 3: pack f32 [K,N] weights into WMMA B fragments ----------------
// frag index = ((nT*KC)+kc)*32+lane ; element e : K = kc*32 + 8*(lane>>4) + e (e<8), kc*32 + 16 + 8*(lane>>4) + (e-8) (e>=8), N = nT*16+(lane&15)
__global__ void pack_w_kernel(const float* __restrict__ W, _Float16* __restrict__ out,
                              int K, int N, int KC, int nTiles) {
    int idx = blockIdx.x * blockDim.x + threadIdx.x;
    int total = nTiles * KC * 32;
    if (idx >= total) return;
    int lane = idx & 31;
    int rest = idx >> 5;
    int kc = rest % KC;
    int nT = rest / KC;
    int n = nT * 16 + (lane & 15);
    size_t base = (size_t)idx * 16;
#pragma unroll
    for (int e = 0; e < 16; ++e) {
        int kk = kc * 32 + ((e < 8) ? ((lane >> 4) * 8 + e) : (16 + (lane >> 4) * 8 + (e - 8)));
        float v = (kk < K && n < N) ? W[(size_t)kk * N + n] : 0.0f;
        out[base + e] = (_Float16)v;
    }
}

// ---------------- kernel 3b: pack f32 [M,K] activations into WMMA A fragments ----------------
// frag index = ((mT*KC)+kc)*32+lane ; same element->K map as load_a_frag.
__global__ void pack_a_kernel(const float* __restrict__ A, _Float16* __restrict__ out,
                              int lda, int K, int KC, int mTiles) {
    int idx = blockIdx.x * blockDim.x + threadIdx.x;
    int total = mTiles * KC * 32;
    if (idx >= total) return;
    int lane = idx & 31;
    int rest = idx >> 5;
    int kc = rest % KC;
    int mT = rest / KC;
    int m  = mT * 16 + (lane & 15);
    int kh = lane >> 4;
    const float* rp = A + (size_t)m * lda;
    size_t base = (size_t)idx * 16;
#pragma unroll
    for (int e = 0; e < 8; ++e) {
        int k1 = kc * 32 + kh * 8 + e;
        int k2 = kc * 32 + 16 + kh * 8 + e;
        out[base + e]     = (_Float16)((k1 < K) ? rp[k1] : 0.0f);
        out[base + 8 + e] = (_Float16)((k2 < K) ? rp[k2] : 0.0f);
    }
}

// ---------------- kernel 4: macro encoder (t=16), one block (4 waves) per sample ----------------
__global__ __launch_bounds__(128)
void macro_encoder_kernel(const float* __restrict__ tokens,
                          float* __restrict__ xflat,
                          const v16h* __restrict__ qkvpk,
                          const v16h* __restrict__ outpk,
                          const v16h* __restrict__ w1pk,
                          const v16h* __restrict__ w2pk,
                          const float* __restrict__ ln1w, const float* __restrict__ ln1b,
                          const float* __restrict__ ln2w, const float* __restrict__ ln2b,
                          const float* __restrict__ b1,  const float* __restrict__ b2) {
    __shared__ float xs[16][64];
    __shared__ float xn[16][64];
    __shared__ float qk[16][192];
    __shared__ float hb[16][256];     // ff hidden; also o-buffer [16][64]
    __shared__ float ps[4][16][16];   // prev (raw) attention scores

    int tid  = threadIdx.x;
    int wave = tid >> 5;
    int lane = tid & 31;
    int b = blockIdx.x;
    int kh = lane >> 4;
    int ln = lane & 15;

#pragma unroll
    for (int i = 0; i < 8; ++i) (&xs[0][0])[tid * 8 + i] = tokens[(size_t)b * 1024 + tid * 8 + i];
#pragma unroll
    for (int i = 0; i < 8; ++i) (&ps[0][0][0])[tid * 8 + i] = 0.0f;
    __syncthreads();

    for (int l = 0; l < LL; ++l) {
        // --- LN1 ---
        if (tid < 16) {
            int t = tid;
            float m = 0.f;
            for (int d = 0; d < 64; ++d) m += xs[t][d];
            m *= (1.0f / 64.0f);
            float var = 0.f;
            for (int d = 0; d < 64; ++d) { float df = xs[t][d] - m; var += df * df; }
            var *= (1.0f / 64.0f);
            float inv = 1.0f / sqrtf(var + EPSLN);
            for (int d = 0; d < 64; ++d)
                xn[t][d] = (xs[t][d] - m) * inv * ln1w[l * 64 + d] + ln1b[l * 64 + d];
        }
        __syncthreads();
        // --- qkv: [16x64] @ [64x192] via WMMA, 12 N tiles / 4 waves ---
        for (int nT = wave; nT < 12; nT += 4) {
            v8f acc = {};
            acc = wmma_tile(&xn[0][0], 64, 64, 2,
                            qkvpk + ((size_t)l * 24 + nT * 2) * 32, lane, acc);
#pragma unroll
            for (int v = 0; v < 8; ++v) qk[v + 8 * kh][nT * 16 + ln] = acc[v];
        }
        __syncthreads();
        // --- attention (VALU): thread = (head, query) ---
        if (tid < 64) {
            int h = tid >> 4, i = tid & 15;
            float sc[16];
            float mx = -3.0e38f;
            for (int j = 0; j < 16; ++j) {
                float d = 0.f;
                for (int dd = 0; dd < 16; ++dd)
                    d += qk[i][h * 16 + dd] * qk[j][64 + h * 16 + dd];
                float v = d * 0.25f + ps[h][i][j];
                ps[h][i][j] = v;             // carry raw scores to next layer
                sc[j] = v;
                mx = fmaxf(mx, v);
            }
            float sum = 0.f;
            for (int j = 0; j < 16; ++j) { sc[j] = expf(sc[j] - mx); sum += sc[j]; }
            float inv = 1.0f / sum;
            for (int dd = 0; dd < 16; ++dd) {
                float o = 0.f;
                for (int j = 0; j < 16; ++j) o += sc[j] * qk[j][128 + h * 16 + dd];
                hb[i][h * 16 + dd] = o * inv;
            }
        }
        __syncthreads();
        // --- out proj + residual: [16x64] @ [64x64], 4 tiles ---
        {
            int nT = wave;
            v8f acc = {};
            acc = wmma_tile(&hb[0][0], 256, 64, 2,
                            outpk + ((size_t)l * 8 + nT * 2) * 32, lane, acc);
            int n = nT * 16 + ln;
#pragma unroll
            for (int v = 0; v < 8; ++v) xs[v + 8 * kh][n] += acc[v];
        }
        __syncthreads();
        // --- LN2 ---
        if (tid < 16) {
            int t = tid;
            float m = 0.f;
            for (int d = 0; d < 64; ++d) m += xs[t][d];
            m *= (1.0f / 64.0f);
            float var = 0.f;
            for (int d = 0; d < 64; ++d) { float df = xs[t][d] - m; var += df * df; }
            var *= (1.0f / 64.0f);
            float inv = 1.0f / sqrtf(var + EPSLN);
            for (int d = 0; d < 64; ++d)
                xn[t][d] = (xs[t][d] - m) * inv * ln2w[l * 64 + d] + ln2b[l * 64 + d];
        }
        __syncthreads();
        // --- ff1 + gelu: [16x64] @ [64x256], 16 tiles ---
        for (int nT = wave; nT < 16; nT += 4) {
            v8f acc = {};
            acc = wmma_tile(&xn[0][0], 64, 64, 2,
                            w1pk + ((size_t)l * 32 + nT * 2) * 32, lane, acc);
            int n = nT * 16 + ln;
            float bv = b1[l * 256 + n];
#pragma unroll
            for (int v = 0; v < 8; ++v) hb[v + 8 * kh][n] = gelu_exact(acc[v] + bv);
        }
        __syncthreads();
        // --- ff2 + bias + residual: [16x256] @ [256x64], 4 tiles ---
        {
            int nT = wave;
            v8f acc = {};
            acc = wmma_tile(&hb[0][0], 256, 256, 8,
                            w2pk + ((size_t)l * 32 + nT * 8) * 32, lane, acc);
            int n = nT * 16 + ln;
            float bv = b2[l * 64 + n];
#pragma unroll
            for (int v = 0; v < 8; ++v) xs[v + 8 * kh][n] += acc[v] + bv;
        }
        __syncthreads();
    }
    // write final x into xflat columns [0,1024)
#pragma unroll
    for (int i = 0; i < 8; ++i)
        xflat[(size_t)b * FLATN + tid * 8 + i] = (&xs[0][0])[tid * 8 + i];
}

// ---------------- kernel 5: continuous projection into xflat cols [1024,1040) ----------------
__global__ void cont_proj_kernel(const float* __restrict__ x_cont,
                                 const float* __restrict__ cont_w,
                                 const float* __restrict__ cont_b,
                                 float* __restrict__ xflat) {
    int idx = blockIdx.x * blockDim.x + threadIdx.x;   // over B*16
    int j = idx & 15;
    int b = idx >> 4;
    float acc = cont_b[j];
    for (int k = 0; k < 16; ++k) acc += x_cont[b * 16 + k] * cont_w[k * 16 + j];
    xflat[(size_t)b * FLATN + NCAT * DD + j] = acc;
}

// ---------------- kernel 6: WMMA GEMM on pre-packed f16 fragments ----------------
// C = act(A @ W + bias). Each wave computes a 4x2 block of 16x16 C tiles
// (register blocking: 4 A-frags + 2 B-frags shared across 8 accumulators,
//  ~21 FLOP per fragment byte). grid = (nTiles/2, mTiles/4/8), block = 256.
template <bool GELU>
__global__ __launch_bounds__(256)
void gemm_wmma_kernel(const v16h* __restrict__ Apk, int KC,
                      const v16h* __restrict__ Wpk,
                      const float* __restrict__ bias,
                      float* __restrict__ C, int ldc) {
    int wave = threadIdx.x >> 5;
    int lane = threadIdx.x & 31;
    int nT0 = blockIdx.x * 2;
    int mT0 = (blockIdx.y * 8 + wave) * 4;

    const v16h* ap0 = Apk + ((size_t)(mT0    ) * KC) * 32 + lane;
    const v16h* ap1 = Apk + ((size_t)(mT0 + 1) * KC) * 32 + lane;
    const v16h* ap2 = Apk + ((size_t)(mT0 + 2) * KC) * 32 + lane;
    const v16h* ap3 = Apk + ((size_t)(mT0 + 3) * KC) * 32 + lane;
    const v16h* bp0 = Wpk + ((size_t)(nT0    ) * KC) * 32 + lane;
    const v16h* bp1 = Wpk + ((size_t)(nT0 + 1) * KC) * 32 + lane;

    v8f acc[4][2] = {};
    for (int kc = 0; kc < KC; ++kc) {
        if (kc + 1 < KC) {   // prefetch next chunk's fragments (global_prefetch_b8)
            __builtin_prefetch(ap0 + (kc + 1) * 32, 0, 0);
            __builtin_prefetch(bp0 + (kc + 1) * 32, 0, 0);
        }
        v16h a0 = ap0[kc * 32];
        v16h a1 = ap1[kc * 32];
        v16h a2 = ap2[kc * 32];
        v16h a3 = ap3[kc * 32];
        v16h b0 = bp0[kc * 32];
        v16h b1 = bp1[kc * 32];
        acc[0][0] = WMMA_F16(a0, b0, acc[0][0]);
        acc[0][1] = WMMA_F16(a0, b1, acc[0][1]);
        acc[1][0] = WMMA_F16(a1, b0, acc[1][0]);
        acc[1][1] = WMMA_F16(a1, b1, acc[1][1]);
        acc[2][0] = WMMA_F16(a2, b0, acc[2][0]);
        acc[2][1] = WMMA_F16(a2, b1, acc[2][1]);
        acc[3][0] = WMMA_F16(a3, b0, acc[3][0]);
        acc[3][1] = WMMA_F16(a3, b1, acc[3][1]);
    }

    int kh = lane >> 4;
    int ln = lane & 15;
    int n0 = nT0 * 16 + ln;
    int n1 = n0 + 16;
    float bv0 = bias[n0];
    float bv1 = bias[n1];
#pragma unroll
    for (int mi = 0; mi < 4; ++mi) {
#pragma unroll
        for (int v = 0; v < 8; ++v) {
            int m = (mT0 + mi) * 16 + v + 8 * kh;
            float c0 = acc[mi][0][v] + bv0;
            float c1 = acc[mi][1][v] + bv1;
            if (GELU) { c0 = gelu_exact(c0); c1 = gelu_exact(c1); }
            C[(size_t)m * ldc + n0] = c0;
            C[(size_t)m * ldc + n1] = c1;
        }
    }
}

// ---------------- kernel 7: final dot  logits[b] = h2[b,:] . w3 + b3 ----------------
__global__ void final_dot_kernel(const float* __restrict__ h2,
                                 const float* __restrict__ w3,
                                 const float* __restrict__ b3,
                                 float* __restrict__ out, int Kdim) {
    __shared__ float red[256];
    int b = blockIdx.x;
    const float* r = h2 + (size_t)b * Kdim;
    float acc = 0.f;
    for (int j = threadIdx.x; j < Kdim; j += 256) acc += r[j] * w3[j];
    red[threadIdx.x] = acc;
    __syncthreads();
    for (int s = 128; s > 0; s >>= 1) {
        if (threadIdx.x < s) red[threadIdx.x] += red[threadIdx.x + s];
        __syncthreads();
    }
    if (threadIdx.x == 0) out[b] = red[0] + b3[0];
}

// ---------------- host ----------------
extern "C" void kernel_launch(void* const* d_in, const int* in_sizes, int n_in,
                              void* d_out, int out_size, void* d_ws, size_t ws_size,
                              hipStream_t stream) {
    const int*   x_cat      = (const int*)  d_in[0];
    const float* x_cont     = (const float*)d_in[1];
    const float* shared_emb = (const float*)d_in[2];
    const float* emb_tables = (const float*)d_in[3];
    const float* mi_qkv  = (const float*)d_in[4];
    const float* mi_out  = (const float*)d_in[5];
    const float* mi_ln1w = (const float*)d_in[6];
    const float* mi_ln1b = (const float*)d_in[7];
    const float* mi_ln2w = (const float*)d_in[8];
    const float* mi_ln2b = (const float*)d_in[9];
    const float* mi_w1   = (const float*)d_in[10];
    const float* mi_b1   = (const float*)d_in[11];
    const float* mi_w2   = (const float*)d_in[12];
    const float* mi_b2   = (const float*)d_in[13];
    const float* ma_qkv  = (const float*)d_in[14];
    const float* ma_out  = (const float*)d_in[15];
    const float* ma_ln1w = (const float*)d_in[16];
    const float* ma_ln1b = (const float*)d_in[17];
    const float* ma_ln2w = (const float*)d_in[18];
    const float* ma_ln2b = (const float*)d_in[19];
    const float* ma_w1   = (const float*)d_in[20];
    const float* ma_b1   = (const float*)d_in[21];
    const float* ma_w2   = (const float*)d_in[22];
    const float* ma_b2   = (const float*)d_in[23];
    const float* cont_w  = (const float*)d_in[24];
    const float* cont_b  = (const float*)d_in[25];
    const float* mlp_w1  = (const float*)d_in[26];
    const float* mlp_b1  = (const float*)d_in[27];
    const float* mlp_w2  = (const float*)d_in[28];
    const float* mlp_b2  = (const float*)d_in[29];
    const float* mlp_w3  = (const float*)d_in[30];
    const float* mlp_b3  = (const float*)d_in[31];
    float* logits = (float*)d_out;

    // ---- workspace carve-up (256B aligned) ----
    size_t off = 0;
    auto take = [&](size_t bytes) {
        size_t cur = off;
        off += (bytes + 255) & ~(size_t)255;
        return cur;
    };
    char* ws = (char*)d_ws;
    float*    tokens = (float*)   (ws + take((size_t)BB * NCAT * DD * 4));
    float*    xflat  = (float*)   (ws + take((size_t)BB * FLATN * 4));
    _Float16* qkvpk  = (_Float16*)(ws + take((size_t)LL * 12 * 2 * 512 * 2));
    _Float16* outpk  = (_Float16*)(ws + take((size_t)LL * 4  * 2 * 512 * 2));
    _Float16* w1pk   = (_Float16*)(ws + take((size_t)LL * 16 * 2 * 512 * 2));
    _Float16* w2pk   = (_Float16*)(ws + take((size_t)LL * 4  * 8 * 512 * 2));
    _Float16* mw1pk  = (_Float16*)(ws + take((size_t)NT1 * KC1 * 512 * 2));
    _Float16* mw2pk  = (_Float16*)(ws + take((size_t)NT2 * KC2 * 512 * 2));
    const int MT_ALL = BB / 16;          // 1024 row tiles over full batch
    const int MT_B   = MB / 16;          // 128 row tiles per MLP batch
    _Float16* xfpk   = (_Float16*)(ws + take((size_t)MT_ALL * KC1 * 512 * 2)); // packed xflat
    float*    h1b    = (float*)   (ws + take((size_t)MB * N1 * 4));
    _Float16* h1pk   = (_Float16*)(ws + take((size_t)MT_B * KC2 * 512 * 2));   // packed h1
    float*    h2b    = (float*)   (ws + take((size_t)MB * N2 * 4));
    (void)ws_size; (void)in_sizes; (void)n_in; (void)out_size;

    // ---- 1. embeddings ----
    build_tokens_kernel<<<(BB * NCAT * DD) / 256, 256, 0, stream>>>(
        x_cat, shared_emb, emb_tables, tokens);

    // ---- 2. micro encoder (rows 0..1 of tokens) ----
    micro_encoder_kernel<<<BB / 8, 256, 0, stream>>>(
        tokens, mi_qkv, mi_out, mi_ln1w, mi_ln1b, mi_ln2w, mi_ln2b,
        mi_w1, mi_b1, mi_w2, mi_b2);

    // ---- 3. pack weights into WMMA B fragments ----
    auto pack = [&](const float* W, _Float16* out, int K, int N, int KC, int nT) {
        int total = nT * KC * 32;
        pack_w_kernel<<<(total + 255) / 256, 256, 0, stream>>>(W, out, K, N, KC, nT);
    };
    for (int l = 0; l < LL; ++l) {
        pack(ma_qkv + (size_t)l * 64 * 192, qkvpk + (size_t)l * 12 * 2 * 512, 64, 192, 2, 12);
        pack(ma_out + (size_t)l * 64 * 64,  outpk + (size_t)l * 4  * 2 * 512, 64, 64,  2, 4);
        pack(ma_w1  + (size_t)l * 64 * 256, w1pk  + (size_t)l * 16 * 2 * 512, 64, 256, 2, 16);
        pack(ma_w2  + (size_t)l * 256 * 64, w2pk  + (size_t)l * 4  * 8 * 512, 256, 64, 8, 4);
    }
    pack(mlp_w1, mw1pk, FLATN, N1, KC1, NT1);
    pack(mlp_w2, mw2pk, N1,    N2, KC2, NT2);

    // ---- 4. macro encoder -> xflat[:, 0:1024) ----
    macro_encoder_kernel<<<BB, 128, 0, stream>>>(
        tokens, xflat, (const v16h*)qkvpk, (const v16h*)outpk,
        (const v16h*)w1pk, (const v16h*)w2pk,
        ma_ln1w, ma_ln1b, ma_ln2w, ma_ln2b, ma_b1, ma_b2);

    // ---- 5. continuous projection -> xflat[:, 1024:1040) ----
    cont_proj_kernel<<<(BB * CONTN) / 256, 256, 0, stream>>>(x_cont, cont_w, cont_b, xflat);

    // ---- 5b. pack all of xflat into f16 A fragments (once, whole batch) ----
    {
        int total = MT_ALL * KC1 * 32;
        pack_a_kernel<<<(total + 255) / 256, 256, 0, stream>>>(
            xflat, xfpk, FLATN, FLATN, KC1, MT_ALL);
    }

    // ---- 6. MLP head, batched over rows ----
    for (int bt = 0; bt < NBATCH; ++bt) {
        const v16h* apk1 = (const v16h*)xfpk + (size_t)bt * MT_B * KC1 * 32;
        dim3 g1(NT1 / 2, MT_B / 4 / 8);   // (130, 4)
        gemm_wmma_kernel<true><<<g1, 256, 0, stream>>>(
            apk1, KC1, (const v16h*)mw1pk, mlp_b1, h1b, N1);

        int totalp = MT_B * KC2 * 32;
        pack_a_kernel<<<(totalp + 255) / 256, 256, 0, stream>>>(
            h1b, h1pk, N1, N1, KC2, MT_B);

        dim3 g2(NT2 / 2, MT_B / 4 / 8);   // (65, 4)
        gemm_wmma_kernel<true><<<g2, 256, 0, stream>>>(
            (const v16h*)h1pk, KC2, (const v16h*)mw2pk, mlp_b2, h2b, N2);

        final_dot_kernel<<<MB, 256, 0, stream>>>(
            h2b, mlp_w3, mlp_b3, logits + (size_t)bt * MB, N2);
    }
}
